// MultiHeadAttention_66279935311990
// MI455X (gfx1250) — matrix operand order repaired and checked by reading, not compiled, not run on hardware
//
#include <hip/hip_runtime.h>
#include <stdint.h>

#define B_ 4
#define T_ 2048
#define C_ 1024
#define H_ 16
#define D_ 64

// softmax scale folded into exp2: (1/sqrt(D)) * log2(e)
#define SCL2E (0.125f * 1.44269504088896340736f)

typedef __attribute__((ext_vector_type(16))) __bf16 v16bf;
typedef __attribute__((ext_vector_type(8)))  float  v8f;

union V16 {
  v16bf v;
  uint4 q[2];
  unsigned short us[16];
};

__device__ __forceinline__ unsigned short f2bf(float f) {
  union { float f; unsigned u; } x; x.f = f;
  unsigned u = x.u + 0x7FFFu + ((x.u >> 16) & 1u);   // round-to-nearest-even
  return (unsigned short)(u >> 16);
}

__device__ __forceinline__ v8f wmma_bf16(v16bf a, v16bf b, v8f c) {
  // D = A(16x32) * B(32x16) + C, fp32 accumulate
  return __builtin_amdgcn_wmma_f32_16x16x32_bf16(false, a, false, b, (short)0, c,
                                                 false, false);
}

// ---- CDNA5 async global->LDS copy (ASYNCcnt-tracked), GV addressing --------
__device__ __forceinline__ void async_copy_b128(unsigned ldsOff, const void* g) {
  asm volatile("global_load_async_to_lds_b128 %0, %1, off"
               :: "v"(ldsOff), "v"(g)
               : "memory");
}
__device__ __forceinline__ void wait_async0() {
  asm volatile("s_wait_asynccnt 0" ::: "memory");
}

// A-operand tile 16(M) x 32(K) bf16 from row-major [*, ld].
// lanes 0-15 row M=lane: elems 0..15 -> K=k0..k0+15;
// lanes 16-31 shift K by +16.
__device__ __forceinline__ v16bf load_a_tile(const unsigned short* base, int m0,
                                             int ld, int k0, int lane) {
  int half = lane >> 4;
  int m = m0 + (lane & 15);
  const unsigned short* p = base + (size_t)m * ld + k0 + half * 16;
  V16 r;
  r.q[0] = *reinterpret_cast<const uint4*>(p);
  r.q[1] = *reinterpret_cast<const uint4*>(p + 8);
  return r.v;
}

// B-operand tile 32(K) x 16(N) bf16 from B^T row-major (column n contiguous).
// lane = column N, 16 contiguous K, lanes 16-31 shift K by +16.
__device__ __forceinline__ v16bf load_b_tile(const unsigned short* base, int n0,
                                             int ld, int k0, int lane) {
  int half = lane >> 4;
  int n = n0 + (lane & 15);
  const unsigned short* p = base + (size_t)n * ld + k0 + half * 16;
  V16 r;
  r.q[0] = *reinterpret_cast<const uint4*>(p);
  r.q[1] = *reinterpret_cast<const uint4*>(p + 8);
  return r.v;
}

// B-operand tile from an LDS slab laid out [64 rows][32 bf16] (64B rows)
__device__ __forceinline__ v16bf lds_b_tile(const unsigned short* slab, int n0,
                                            int lane) {
  int half = lane >> 4;
  int n = n0 + (lane & 15);
  const unsigned short* p = slab + n * 32 + half * 16;
  V16 r;
  r.q[0] = *reinterpret_cast<const uint4*>(p);
  r.q[1] = *reinterpret_cast<const uint4*>(p + 8);
  return r.v;
}

// ---------------------------------------------------------------- conversions

__global__ void cvt_x_kernel(const float* __restrict__ x,
                             unsigned short* __restrict__ xb, int n) {
  int i = (blockIdx.x * blockDim.x + threadIdx.x) * 4;
  if (i >= n) return;
  float4 f = *reinterpret_cast<const float4*>(x + i);
  ushort4 o;
  o.x = f2bf(f.x); o.y = f2bf(f.y); o.z = f2bf(f.z); o.w = f2bf(f.w);
  *reinterpret_cast<ushort4*>(xb + i) = o;
}

// Wq/Wk/Wv [H][C][D] fp32 -> WqkvT [3][H][D][C] bf16
__global__ void cvt_wqkv_kernel(const float* __restrict__ Wq,
                                const float* __restrict__ Wk,
                                const float* __restrict__ Wv,
                                unsigned short* __restrict__ outT) {
  int idx = blockIdx.x * blockDim.x + threadIdx.x;   // over 3*H*D*C
  int c = idx % C_;
  int t = idx / C_;
  int d = t % D_; t /= D_;
  int h = t % H_; t /= H_;
  const float* W = (t == 0) ? Wq : (t == 1) ? Wk : Wv;
  outT[idx] = f2bf(W[((size_t)h * C_ + c) * D_ + d]);
}

// Wp [HD][C] fp32 -> WpT [C][HD] bf16
__global__ void cvt_wp_kernel(const float* __restrict__ Wp,
                              unsigned short* __restrict__ WpT) {
  int idx = blockIdx.x * blockDim.x + threadIdx.x;   // over C*HD
  int k = idx % (H_ * D_);
  int n = idx / (H_ * D_);
  WpT[idx] = f2bf(Wp[(size_t)k * C_ + n]);
}

// ---------------------------------------------------------------- QKV GEMM
// Block = 128 threads (4 waves), all share one (b,h,s): the 64xC weight matrix
// is async-staged to LDS (double buffered, 4KB slabs) and reused by all waves.
// Each wave computes 32 t-rows x 64 d-cols (8 accumulator tiles).

__global__ void __launch_bounds__(128)
qkv_gemm_kernel(const unsigned short* __restrict__ xb,
                const unsigned short* __restrict__ WqkvT,
                unsigned short* __restrict__ q,
                unsigned short* __restrict__ k,
                unsigned short* __restrict__ vT) {
  __shared__ __align__(16) unsigned short Bs[2][64 * 32];   // 2 x 4KB

  int tid = threadIdx.x;
  int lane = tid & 31;
  int wave = tid >> 5;

  int bx = blockIdx.x;                 // (b*H + h)*3 + s
  int s = bx % 3; bx /= 3;
  int h = bx % H_;
  int b = bx / H_;
  int t0 = blockIdx.y * 128 + wave * 32;

  const unsigned short* A  = xb + (size_t)b * T_ * C_;
  const unsigned short* Bw = WqkvT + (size_t)(s * H_ + h) * D_ * C_;

  // slab copy plan: 256 16B units/step; thread handles units tid and tid+128
  int u0 = tid, u1 = tid + 128;
  const unsigned short* g0 = Bw + (size_t)(u0 >> 2) * C_ + (u0 & 3) * 8;
  const unsigned short* g1 = Bw + (size_t)(u1 >> 2) * C_ + (u1 & 3) * 8;
  unsigned lb = (unsigned)(size_t)(&Bs[0][0]);
  unsigned lo0 = (unsigned)u0 * 16, lo1 = (unsigned)u1 * 16;

  // prologue: stage slab 0
  async_copy_b128(lb + lo0, g0);
  async_copy_b128(lb + lo1, g1);
  wait_async0();
  __syncthreads();

  v8f acc[8] = {};   // [rowblk 0..1][ntile 0..3]
  v16bf a0 = load_a_tile(A, t0,      C_, 0, lane);
  v16bf a1 = load_a_tile(A, t0 + 16, C_, 0, lane);

  for (int step = 0; step < C_ / 32; ++step) {
    // async-stage next weight slab while this step computes
    if (step + 1 < C_ / 32) {
      unsigned nxt = lb + (unsigned)((step + 1) & 1) * 4096;
      async_copy_b128(nxt + lo0, g0 + (step + 1) * 32);
      async_copy_b128(nxt + lo1, g1 + (step + 1) * 32);
    }
    // register-prefetch next A tiles
    v16bf na0, na1;
    if (step + 1 < C_ / 32) {
      na0 = load_a_tile(A, t0,      C_, (step + 1) * 32, lane);
      na1 = load_a_tile(A, t0 + 16, C_, (step + 1) * 32, lane);
    }

    const unsigned short* slab = &Bs[step & 1][0];
#pragma unroll
    for (int nt = 0; nt < 4; ++nt) {
      v16bf bt = lds_b_tile(slab, nt * 16, lane);
      acc[nt]     = wmma_bf16(a0, bt, acc[nt]);
      acc[4 + nt] = wmma_bf16(a1, bt, acc[4 + nt]);
    }

    wait_async0();      // next slab resident
    __syncthreads();    // all waves done reading current slab
    a0 = na0; a1 = na1;
  }

  int half = lane >> 4;
  int nl = lane & 15;
  size_t bh = (size_t)(b * H_ + h);
  if (s == 2) {
    unsigned short* dst = vT + bh * D_ * T_;   // [d][t]
#pragma unroll
    for (int rb = 0; rb < 2; ++rb) {
#pragma unroll
      for (int r = 0; r < 8; ++r) {
        int t = t0 + rb * 16 + r + half * 8;
        dst[(size_t)( 0 + nl) * T_ + t] = f2bf(acc[rb * 4 + 0][r]);
        dst[(size_t)(16 + nl) * T_ + t] = f2bf(acc[rb * 4 + 1][r]);
        dst[(size_t)(32 + nl) * T_ + t] = f2bf(acc[rb * 4 + 2][r]);
        dst[(size_t)(48 + nl) * T_ + t] = f2bf(acc[rb * 4 + 3][r]);
      }
    }
  } else {
    unsigned short* dst = (s == 0 ? q : k) + bh * T_ * D_;  // [t][d]
#pragma unroll
    for (int rb = 0; rb < 2; ++rb) {
#pragma unroll
      for (int r = 0; r < 8; ++r) {
        int t = t0 + rb * 16 + r + half * 8;
        dst[(size_t)t * D_ +  0 + nl] = f2bf(acc[rb * 4 + 0][r]);
        dst[(size_t)t * D_ + 16 + nl] = f2bf(acc[rb * 4 + 1][r]);
        dst[(size_t)t * D_ + 32 + nl] = f2bf(acc[rb * 4 + 2][r]);
        dst[(size_t)t * D_ + 48 + nl] = f2bf(acc[rb * 4 + 3][r]);
      }
    }
  }
}

// ---------------------------------------------------------------- flash attention
// One wave owns one 16-row t-tile of one (b,h). Online softmax over s chunks of 32.
// V tiles + next-chunk K tiles are issued right after the score WMMAs so the
// softmax VALU/trans work shadows their latency.

__global__ void __launch_bounds__(128)
flash_attn_kernel(const unsigned short* __restrict__ q,
                  const unsigned short* __restrict__ kk,
                  const unsigned short* __restrict__ vT,
                  unsigned short* __restrict__ attn) {
  __shared__ __align__(16) unsigned short ldsP[4][16 * 32];  // per-wave P staging

  int lane = threadIdx.x & 31;
  int wave = threadIdx.x >> 5;
  int gw = blockIdx.x * 4 + wave;           // over B*H*(T/16)
  int ttile = gw % (T_ / 16);
  int bh = gw / (T_ / 16);
  int h = bh % H_;
  int b = bh / H_;
  int t0 = ttile * 16;

  const unsigned short* Q = q  + (size_t)bh * T_ * D_;
  const unsigned short* K = kk + (size_t)bh * T_ * D_;
  const unsigned short* V = vT + (size_t)bh * D_ * T_;

  int half = lane >> 4;
  int nl = lane & 15;

  v16bf qa0 = load_a_tile(Q, t0, D_, 0, lane);
  v16bf qa1 = load_a_tile(Q, t0, D_, 32, lane);

  v8f o0 = {}, o1 = {}, o2 = {}, o3 = {};
  float m_run[8], l_run[8];
#pragma unroll
  for (int r = 0; r < 8; ++r) { m_run[r] = -3.0e38f; l_run[r] = 0.0f; }

  unsigned short* P = &ldsP[wave][0];

  int nchunk = (t0 + 15) / 32 + 1;          // causal: cover s <= t0+15

  v16bf kb0 = load_b_tile(K,  0, D_,  0, lane);
  v16bf kb1 = load_b_tile(K,  0, D_, 32, lane);
  v16bf kb2 = load_b_tile(K, 16, D_,  0, lane);
  v16bf kb3 = load_b_tile(K, 16, D_, 32, lane);

  for (int c = 0; c < nchunk; ++c) {
    int sbase = c * 32;

    v8f sA = {}, sB = {};
    sA = wmma_bf16(qa0, kb0, sA);
    sA = wmma_bf16(qa1, kb1, sA);
    sB = wmma_bf16(qa0, kb2, sB);
    sB = wmma_bf16(qa1, kb3, sB);

    // issue V loads for this chunk now; consumed only after softmax
    v16bf vb0 = load_b_tile(V,  0, T_, sbase, lane);
    v16bf vb1 = load_b_tile(V, 16, T_, sbase, lane);
    v16bf vb2 = load_b_tile(V, 32, T_, sbase, lane);
    v16bf vb3 = load_b_tile(V, 48, T_, sbase, lane);

    // prefetch next chunk's K tiles under the softmax shadow
    if (c + 1 < nchunk) {
      kb0 = load_b_tile(K, sbase + 32, D_,  0, lane);
      kb1 = load_b_tile(K, sbase + 32, D_, 32, lane);
      kb2 = load_b_tile(K, sbase + 48, D_,  0, lane);
      kb3 = load_b_tile(K, sbase + 48, D_, 32, lane);
    }

    if (sbase + 31 > t0) {                  // diagonal chunk: causal mask
      int sa = sbase + nl;
      int sb = sbase + 16 + nl;
#pragma unroll
      for (int r = 0; r < 8; ++r) {
        int t = t0 + r + half * 8;
        if (sa > t) sA[r] = -3.0e38f;
        if (sb > t) sB[r] = -3.0e38f;
      }
    }

    float alpha[8];
#pragma unroll
    for (int r = 0; r < 8; ++r) {
      float cm = fmaxf(sA[r], sB[r]);       // row reduce over s within half-wave
      cm = fmaxf(cm, __shfl_xor(cm, 1, 32));
      cm = fmaxf(cm, __shfl_xor(cm, 2, 32));
      cm = fmaxf(cm, __shfl_xor(cm, 4, 32));
      cm = fmaxf(cm, __shfl_xor(cm, 8, 32));
      float m_new = fmaxf(m_run[r], cm);
      alpha[r] = __builtin_exp2f((m_run[r] - m_new) * SCL2E);
      float pa = __builtin_exp2f((sA[r] - m_new) * SCL2E);
      float pb = __builtin_exp2f((sB[r] - m_new) * SCL2E);
      sA[r] = pa; sB[r] = pb;
      float rs = pa + pb;
      rs += __shfl_xor(rs, 1, 32);
      rs += __shfl_xor(rs, 2, 32);
      rs += __shfl_xor(rs, 4, 32);
      rs += __shfl_xor(rs, 8, 32);
      l_run[r] = l_run[r] * alpha[r] + rs;
      m_run[r] = m_new;
    }

#pragma unroll
    for (int r = 0; r < 8; ++r) {           // rescale running output
      o0[r] *= alpha[r]; o1[r] *= alpha[r];
      o2[r] *= alpha[r]; o3[r] *= alpha[r];
    }

    // transpose P: D-layout -> A-layout through wave-private LDS
#pragma unroll
    for (int r = 0; r < 8; ++r) {
      int t = r + half * 8;
      P[t * 32 + nl]      = f2bf(sA[r]);
      P[t * 32 + 16 + nl] = f2bf(sB[r]);
    }
    __asm__ volatile("s_wait_dscnt 0" ::: "memory");  // same-wave DS ordering

    V16 pv;
    {
      const unsigned short* pr = P + nl * 32 + half * 16;
      pv.q[0] = *reinterpret_cast<const uint4*>(pr);
      pv.q[1] = *reinterpret_cast<const uint4*>(pr + 8);
    }

    o0 = wmma_bf16(pv.v, vb0, o0);
    o1 = wmma_bf16(pv.v, vb1, o1);
    o2 = wmma_bf16(pv.v, vb2, o2);
    o3 = wmma_bf16(pv.v, vb3, o3);
  }

  // normalize and store concat-heads output: attn[(b*T+t)][h*D + d] bf16
  unsigned short* dst = attn + (size_t)b * T_ * (H_ * D_) + h * D_;
#pragma unroll
  for (int r = 0; r < 8; ++r) {
    int t = t0 + r + half * 8;
    float inv = 1.0f / l_run[r];
    unsigned short* row = dst + (size_t)t * (H_ * D_);
    row[ 0 + nl] = f2bf(o0[r] * inv);
    row[16 + nl] = f2bf(o1[r] * inv);
    row[32 + nl] = f2bf(o2[r] * inv);
    row[48 + nl] = f2bf(o3[r] * inv);
  }
}

// ---------------------------------------------------------------- output projection
// Same async-staged structure as QKV: all 4 waves share the 64-column Wp slab.

__global__ void __launch_bounds__(128)
proj_gemm_kernel(const unsigned short* __restrict__ attn,
                 const unsigned short* __restrict__ WpT,
                 const float* __restrict__ bp,
                 float* __restrict__ out) {
  __shared__ __align__(16) unsigned short Bs[2][64 * 32];   // 2 x 4KB

  int tid = threadIdx.x;
  int lane = tid & 31;
  int wave = tid >> 5;
  int m0 = blockIdx.x * 128 + wave * 32;
  int n0 = blockIdx.y * 64;
  const int K_ = H_ * D_;

  int u0 = tid, u1 = tid + 128;
  const unsigned short* g0 = WpT + (size_t)(n0 + (u0 >> 2)) * K_ + (u0 & 3) * 8;
  const unsigned short* g1 = WpT + (size_t)(n0 + (u1 >> 2)) * K_ + (u1 & 3) * 8;
  unsigned lb = (unsigned)(size_t)(&Bs[0][0]);
  unsigned lo0 = (unsigned)u0 * 16, lo1 = (unsigned)u1 * 16;

  async_copy_b128(lb + lo0, g0);
  async_copy_b128(lb + lo1, g1);
  wait_async0();
  __syncthreads();

  v8f acc[8] = {};
  v16bf a0 = load_a_tile(attn, m0,      K_, 0, lane);
  v16bf a1 = load_a_tile(attn, m0 + 16, K_, 0, lane);

  for (int step = 0; step < K_ / 32; ++step) {
    if (step + 1 < K_ / 32) {
      unsigned nxt = lb + (unsigned)((step + 1) & 1) * 4096;
      async_copy_b128(nxt + lo0, g0 + (step + 1) * 32);
      async_copy_b128(nxt + lo1, g1 + (step + 1) * 32);
    }
    v16bf na0, na1;
    if (step + 1 < K_ / 32) {
      na0 = load_a_tile(attn, m0,      K_, (step + 1) * 32, lane);
      na1 = load_a_tile(attn, m0 + 16, K_, (step + 1) * 32, lane);
    }

    const unsigned short* slab = &Bs[step & 1][0];
#pragma unroll
    for (int nt = 0; nt < 4; ++nt) {
      v16bf bt = lds_b_tile(slab, nt * 16, lane);
      acc[nt]     = wmma_bf16(a0, bt, acc[nt]);
      acc[4 + nt] = wmma_bf16(a1, bt, acc[4 + nt]);
    }

    wait_async0();
    __syncthreads();
    a0 = na0; a1 = na1;
  }

  int half = lane >> 4;
  int nl = lane & 15;
  float b0 = bp[n0 + nl],      b1 = bp[n0 + 16 + nl];
  float b2 = bp[n0 + 32 + nl], b3 = bp[n0 + 48 + nl];
#pragma unroll
  for (int rb = 0; rb < 2; ++rb) {
#pragma unroll
    for (int r = 0; r < 8; ++r) {
      int m = m0 + rb * 16 + r + half * 8;
      float* row = out + (size_t)m * C_;
      row[n0 +  0 + nl] = acc[rb * 4 + 0][r] + b0;
      row[n0 + 16 + nl] = acc[rb * 4 + 1][r] + b1;
      row[n0 + 32 + nl] = acc[rb * 4 + 2][r] + b2;
      row[n0 + 48 + nl] = acc[rb * 4 + 3][r] + b3;
    }
  }
}

// ---------------------------------------------------------------- launch

extern "C" void kernel_launch(void* const* d_in, const int* in_sizes, int n_in,
                              void* d_out, int out_size, void* d_ws, size_t ws_size,
                              hipStream_t stream) {
  const float* x  = (const float*)d_in[0];
  const float* Wq = (const float*)d_in[1];
  const float* Wk = (const float*)d_in[2];
  const float* Wv = (const float*)d_in[3];
  const float* Wp = (const float*)d_in[4];
  const float* bp = (const float*)d_in[5];
  float* out = (float*)d_out;

  char* ws = (char*)d_ws;
  // region 0 (16 MB) holds xb during QKV, then is reused for attn output
  unsigned short* xb    = (unsigned short*)(ws);
  unsigned short* attn  = (unsigned short*)(ws);
  unsigned short* q     = (unsigned short*)(ws + (16ull << 20));
  unsigned short* k     = (unsigned short*)(ws + (32ull << 20));
  unsigned short* vT    = (unsigned short*)(ws + (48ull << 20));
  unsigned short* WqkvT = (unsigned short*)(ws + (64ull << 20));   // 6 MB
  unsigned short* WpT   = (unsigned short*)(ws + (70ull << 20));   // 2 MB

  cvt_x_kernel<<<(B_ * T_ * C_) / 1024, 256, 0, stream>>>(x, xb, B_ * T_ * C_);
  cvt_wqkv_kernel<<<(3 * H_ * D_ * C_) / 256, 256, 0, stream>>>(Wq, Wk, Wv, WqkvT);
  cvt_wp_kernel<<<(C_ * H_ * D_) / 256, 256, 0, stream>>>(Wp, WpT);

  dim3 gq(B_ * H_ * 3, T_ / 128);
  qkv_gemm_kernel<<<gq, 128, 0, stream>>>(xb, WqkvT, q, k, vT);

  flash_attn_kernel<<<(B_ * H_ * (T_ / 16)) / 4, 128, 0, stream>>>(q, k, vT, attn);

  dim3 gp((B_ * T_) / 128, C_ / 64);
  proj_gemm_kernel<<<gp, 128, 0, stream>>>(attn, WpT, bp, out);
}
